// Interaction_Block_3143916061060
// MI455X (gfx1250) — compile-verified, matrix operand order checked by reading, not run on hardware
//
#include <hip/hip_runtime.h>
#include <hip/hip_bf16.h>

// MI455X / gfx1250 implementation.
// All 128-wide MLP layers run through v_wmma_f32_16x16x32_f16 (f32 accum).
// Weights are pre-transposed to f16 once (wprep), then each GEMM block stages
// its 128-K weight panel into LDS with GLOBAL_LOAD_ASYNC_TO_LDS_B128
// (ASYNCcnt path). Bias + ReLU + LayerNorm fused in the WMMA epilogue.

#define N_NODES 100000
#define N_EDGES 400000
#define LATENT  128

typedef __attribute__((ext_vector_type(16))) _Float16 v16h;
typedef __attribute__((ext_vector_type(8)))  float    v8f;

union Frag {
  v16h v;
  uint4 q[2];
  _Float16 h[16];
};

// One-time weight prep: Wt[n*K + k] = (f16) W[k*128 + n]   (transposed, f16)
__global__ __launch_bounds__(256) void wprep_kernel(const float* __restrict__ W,
                                                    _Float16* __restrict__ Wt,
                                                    int K) {
  int i = blockIdx.x * 256 + threadIdx.x;
  if (i >= 128 * K) return;
  int n = i / K, k = i - n * K;
  Wt[i] = (_Float16)W[(size_t)k * 128 + n];
}

// ---------------------------------------------------------------------------
// Generic fused GEMM: Y = epilogue(X[M,K] @ W[K,128] + b), X,W,Y16 in f16.
// 256 threads = 8 waves; each wave computes a 16x128 strip of Y.
// Wt is the transposed f16 weight [128][K] produced by wprep_kernel.
// ---------------------------------------------------------------------------
template<int K, bool RELU, bool LN, bool OUT16, bool OUT32>
__global__ __launch_bounds__(256) void gemm128_kernel(
    const _Float16* __restrict__ X, const _Float16* __restrict__ Wt,
    const float* __restrict__ bias,
    const float* __restrict__ ln_g, const float* __restrict__ ln_b,
    _Float16* __restrict__ Y16, float* __restrict__ Y32, int M)
{
  __shared__ _Float16 sW[128 * 128];   // transposed weight panel: sW[n*128 + kk]

  const int tid   = threadIdx.x;
  const int lane  = tid & 31;
  const int wave  = tid >> 5;
  const int n16   = lane & 15;          // col within 16-tile / A row within tile
  const int kh    = (lane >> 4) * 8;    // K sub-offset per half-wave
  const int rowBase = blockIdx.x * 128 + wave * 16;
  const int rowA    = rowBase + n16;
  const uint32_t sWbase = (uint32_t)(uintptr_t)(&sW[0]);

  v8f acc[8] = {};

  for (int panel = 0; panel < K; panel += 128) {
    __syncthreads();
    // Async-stage the 32KB transposed panel: sW[n*128+kk] = Wt[n*K+panel+kk].
    // 2048 x 16B transfers via the CDNA5 ASYNCcnt direct-to-LDS path.
    for (int c = tid; c < 2048; c += 256) {
      int n = c >> 4, part = c & 15;
      const _Float16* src = Wt + (size_t)n * K + panel + part * 8;
      uint32_t dst = sWbase + (uint32_t)(n * 256 + part * 16);
      asm volatile("global_load_async_to_lds_b128 %0, %1, off"
                   :: "v"(dst), "v"((unsigned long long)(uintptr_t)src)
                   : "memory");
    }
    asm volatile("s_wait_asynccnt 0x0" ::: "memory");
    __syncthreads();

    if (panel + 128 < K && rowA < M)   // gfx1250 global_prefetch_b8 path
      __builtin_prefetch(X + (size_t)rowA * K + panel + 128, 0, 1);

    #pragma unroll
    for (int kb = 0; kb < 128; kb += 32) {
      Frag a;
      if (rowA < M) {
        const _Float16* ap = X + (size_t)rowA * K + panel + kb + kh;
        a.q[0] = *(const uint4*)(ap);        // K = base+kh+0..7
        a.q[1] = *(const uint4*)(ap + 16);   // K = base+16+kh+0..7
      } else {
        a.q[0] = make_uint4(0, 0, 0, 0);
        a.q[1] = make_uint4(0, 0, 0, 0);
      }
      #pragma unroll
      for (int nt = 0; nt < 8; ++nt) {
        Frag b;
        const _Float16* bp = &sW[(nt * 16 + n16) * 128 + kb + kh];
        b.q[0] = *(const uint4*)(bp);
        b.q[1] = *(const uint4*)(bp + 16);
        acc[nt] = __builtin_amdgcn_wmma_f32_16x16x32_f16(
            false, a.v, false, b.v, (short)0, acc[nt], false, false);
      }
    }
  }

  // ---- epilogue: bias (+ReLU) (+LayerNorm over 128 cols) ----
  float colb[8], colg[8], colbeta[8];
  #pragma unroll
  for (int nt = 0; nt < 8; ++nt) {
    int col = nt * 16 + n16;
    colb[nt] = bias[col];
    if constexpr (LN) { colg[nt] = ln_g[col]; colbeta[nt] = ln_b[col]; }
  }
  #pragma unroll
  for (int nt = 0; nt < 8; ++nt)
    #pragma unroll
    for (int j = 0; j < 8; ++j) {
      float v = acc[nt][j] + colb[nt];
      if constexpr (RELU) v = fmaxf(v, 0.0f);
      acc[nt][j] = v;
    }

  float mean[8], rstd[8];
  if constexpr (LN) {
    #pragma unroll
    for (int j = 0; j < 8; ++j) {
      float s = 0.f, s2 = 0.f;
      #pragma unroll
      for (int nt = 0; nt < 8; ++nt) { float v = acc[nt][j]; s += v; s2 += v * v; }
      #pragma unroll
      for (int m = 1; m < 16; m <<= 1) {
        s  += __shfl_xor(s,  m, 32);
        s2 += __shfl_xor(s2, m, 32);
      }
      float mu  = s * (1.0f / 128.0f);
      float var = s2 * (1.0f / 128.0f) - mu * mu;
      mean[j] = mu;
      rstd[j] = rsqrtf(var + 1e-5f);
    }
  }

  const int rlo = (lane >> 4) * 8;
  #pragma unroll
  for (int j = 0; j < 8; ++j) {
    int row = rowBase + rlo + j;
    if (row >= M) continue;
    #pragma unroll
    for (int nt = 0; nt < 8; ++nt) {
      int col = nt * 16 + n16;
      float v = acc[nt][j];
      if constexpr (LN) v = (v - mean[j]) * rstd[j] * colg[nt] + colbeta[nt];
      if constexpr (OUT16) Y16[(size_t)row * 128 + col] = (_Float16)v;
      if constexpr (OUT32) Y32[(size_t)row * 128 + col] = v;
    }
  }
}

// ---------------------------------------------------------------------------
// Edge front: basis projections, feature build, tiny-K first layers (6->128,
// 2->128) of edge_feat (x2) and edge MLPs, ReLU, store f16 hidden states.
// ---------------------------------------------------------------------------
__global__ __launch_bounds__(256) void edge_front_kernel(
    const float* __restrict__ edge_dx, const float* __restrict__ edge_attr,
    const float* __restrict__ va, const float* __restrict__ vb, const float* __restrict__ vc,
    const float* __restrict__ svt, const float* __restrict__ svtm1,
    const float* __restrict__ rvt, const float* __restrict__ rvtm1,
    const float* __restrict__ efW1, const float* __restrict__ efb1,
    const float* __restrict__ eW1,  const float* __restrict__ eb1,
    _Float16* __restrict__ hs, _Float16* __restrict__ hr, _Float16* __restrict__ he)
{
  __shared__ float sWef[6 * 128];
  __shared__ float sbef[128];
  __shared__ float sWe[2 * 128];
  __shared__ float sbe[128];
  int tid = threadIdx.x;
  for (int i = tid; i < 6 * 128; i += 256) sWef[i] = efW1[i];
  for (int i = tid; i < 2 * 128; i += 256) sWe[i]  = eW1[i];
  for (int i = tid; i < 128; i += 256) { sbef[i] = efb1[i]; sbe[i] = eb1[i]; }
  __syncthreads();

  int e = blockIdx.x * 256 + tid;
  if (e >= N_EDGES) return;

  float B[3][3];
  #pragma unroll
  for (int k = 0; k < 3; ++k) {
    B[0][k] = va[(size_t)e * 3 + k];
    B[1][k] = vb[(size_t)e * 3 + k];
    B[2][k] = vc[(size_t)e * 3 + k];
  }
  float vs[3], vs1[3], vr[3], vr1[3], dx[3];
  #pragma unroll
  for (int k = 0; k < 3; ++k) {
    vs[k]  = svt  [(size_t)e * 3 + k];
    vs1[k] = svtm1[(size_t)e * 3 + k];
    vr[k]  = rvt  [(size_t)e * 3 + k];
    vr1[k] = rvtm1[(size_t)e * 3 + k];
    dx[k]  = edge_dx[(size_t)e * 3 + k];
  }
  float fs[6], fr[6];
  #pragma unroll
  for (int r = 0; r < 3; ++r) {
    fs[r]     = B[r][0] * vs[0]  + B[r][1] * vs[1]  + B[r][2] * vs[2];
    fs[3 + r] = B[r][0] * vs1[0] + B[r][1] * vs1[1] + B[r][2] * vs1[2];
    fr[r]     = -(B[r][0] * vr[0]  + B[r][1] * vr[1]  + B[r][2] * vr[2]);
    fr[3 + r] = -(B[r][0] * vr1[0] + B[r][1] * vr1[1] + B[r][2] * vr1[2]);
  }
  float fe0 = sqrtf(dx[0] * dx[0] + dx[1] * dx[1] + dx[2] * dx[2]);
  float fe1 = edge_attr[e];

  for (int j = 0; j < 128; ++j) {
    float as = sbef[j], ar = sbef[j];
    #pragma unroll
    for (int i = 0; i < 6; ++i) {
      float w = sWef[i * 128 + j];
      as += fs[i] * w;
      ar += fr[i] * w;
    }
    float ae = sbe[j] + fe0 * sWe[j] + fe1 * sWe[128 + j];
    hs[(size_t)e * 128 + j] = (_Float16)fmaxf(as, 0.0f);
    hr[(size_t)e * 128 + j] = (_Float16)fmaxf(ar, 0.0f);
    he[(size_t)e * 128 + j] = (_Float16)fmaxf(ae, 0.0f);
  }
}

// msg_input[E,384] = [s_lat+r_lat | node_latent[s]+node_latent[r] | e_lat] (f16)
__global__ __launch_bounds__(256) void msg_kernel(
    const _Float16* __restrict__ slat, const _Float16* __restrict__ rlat,
    const _Float16* __restrict__ elat, const float* __restrict__ node_latent,
    const int* __restrict__ edge_index, _Float16* __restrict__ msg)
{
  long long idx = (long long)blockIdx.x * 256 + threadIdx.x;
  if (idx >= (long long)N_EDGES * 128) return;
  int e = (int)(idx >> 7);
  int j = (int)(idx & 127);
  int s = edge_index[e];
  int r = edge_index[N_EDGES + e];
  size_t mrow = (size_t)e * 384;
  msg[mrow + j]       = (_Float16)((float)slat[idx] + (float)rlat[idx]);
  msg[mrow + 128 + j] = (_Float16)(node_latent[(size_t)s * 128 + j] +
                                   node_latent[(size_t)r * 128 + j]);
  msg[mrow + 256 + j] = elat[idx];
}

// i1 final layer (128->3) + fij = c0*a + c1*b + c2*c, scatter-add to receivers
__global__ __launch_bounds__(256) void fij_scatter_kernel(
    const _Float16* __restrict__ h, const float* __restrict__ W3,
    const float* __restrict__ b3,
    const float* __restrict__ va, const float* __restrict__ vb,
    const float* __restrict__ vc,
    const int* __restrict__ edge_index, float* __restrict__ fij_out)
{
  __shared__ float sW[128 * 3];
  __shared__ float sb[3];
  int tid = threadIdx.x;
  for (int i = tid; i < 128 * 3; i += 256) sW[i] = W3[i];
  if (tid < 3) sb[tid] = b3[tid];
  __syncthreads();

  int e = blockIdx.x * 256 + tid;
  if (e >= N_EDGES) return;
  float c0 = sb[0], c1 = sb[1], c2 = sb[2];
  for (int k = 0; k < 128; ++k) {
    float x = (float)h[(size_t)e * 128 + k];
    c0 += x * sW[k * 3 + 0];
    c1 += x * sW[k * 3 + 1];
    c2 += x * sW[k * 3 + 2];
  }
  int r = edge_index[N_EDGES + e];
  #pragma unroll
  for (int d = 0; d < 3; ++d) {
    float f = c0 * va[(size_t)e * 3 + d] + c1 * vb[(size_t)e * 3 + d] +
              c2 * vc[(size_t)e * 3 + d];
    atomicAdd(&fij_out[(size_t)r * 3 + d], f);
  }
}

__global__ __launch_bounds__(256) void zero_kernel(float* __restrict__ p, long long n) {
  long long i = (long long)blockIdx.x * 256 + threadIdx.x;
  if (i < n) p[i] = 0.0f;
}

__global__ __launch_bounds__(256) void cvt16_kernel(const float* __restrict__ x,
                                                    _Float16* __restrict__ y,
                                                    long long n) {
  long long i = (long long)blockIdx.x * 256 + threadIdx.x;
  if (i < n) y[i] = (_Float16)x[i];
}

// node finalize: m = hm@Wm3+bm3 ; fext = hf@Wf3+bf3 ; resid = m*acc + fij - fext
__global__ __launch_bounds__(256) void node_final_kernel(
    const _Float16* __restrict__ hm, const _Float16* __restrict__ hf,
    const float* __restrict__ mW3, const float* __restrict__ mb3,
    const float* __restrict__ fW3, const float* __restrict__ fb3,
    const float* __restrict__ node_acc, const float* __restrict__ fij,
    float* __restrict__ resid)
{
  __shared__ float sWm[128];
  __shared__ float sWf[128 * 3];
  __shared__ float sbm;
  __shared__ float sbf[3];
  int tid = threadIdx.x;
  for (int i = tid; i < 128; i += 256) sWm[i] = mW3[i];
  for (int i = tid; i < 128 * 3; i += 256) sWf[i] = fW3[i];
  if (tid == 0) sbm = mb3[0];
  if (tid < 3) sbf[tid] = fb3[tid];
  __syncthreads();

  int n = blockIdx.x * 256 + tid;
  if (n >= N_NODES) return;
  float m = sbm;
  float f0 = sbf[0], f1 = sbf[1], f2 = sbf[2];
  for (int k = 0; k < 128; ++k) {
    float xm = (float)hm[(size_t)n * 128 + k];
    float xf = (float)hf[(size_t)n * 128 + k];
    m  += xm * sWm[k];
    f0 += xf * sWf[k * 3 + 0];
    f1 += xf * sWf[k * 3 + 1];
    f2 += xf * sWf[k * 3 + 2];
  }
  float fe[3] = {f0, f1, f2};
  #pragma unroll
  for (int d = 0; d < 3; ++d)
    resid[(size_t)n * 3 + d] =
        m * node_acc[(size_t)n * 3 + d] + fij[(size_t)n * 3 + d] - fe[d];
}

// ---------------------------------------------------------------------------
extern "C" void kernel_launch(void* const* d_in, const int* in_sizes, int n_in,
                              void* d_out, int out_size, void* d_ws, size_t ws_size,
                              hipStream_t stream) {
  (void)in_sizes; (void)n_in; (void)out_size; (void)ws_size;

  // ---- inputs (setup_inputs insertion order) ----
  const int*   edge_index = (const int*)d_in[0];
  const float* edge_dx    = (const float*)d_in[3];
  const float* edge_attr  = (const float*)d_in[4];
  const float* va         = (const float*)d_in[5];
  const float* vb         = (const float*)d_in[6];
  const float* vc         = (const float*)d_in[7];
  const float* svt        = (const float*)d_in[8];
  const float* svtm1      = (const float*)d_in[9];
  const float* rvt        = (const float*)d_in[10];
  const float* rvtm1      = (const float*)d_in[11];
  const float* node_acc   = (const float*)d_in[12];
  const float* node_latent= (const float*)d_in[13];

  // params flattened: edge_feat, edge, inter (W1,b1,W2,b2,W3,b3,g,beta), then
  // i1, m, fext (W1,b1,W2,b2,W3,b3)
  const float* P[42];
  for (int i = 0; i < 42; ++i) P[i] = (const float*)d_in[14 + i];
  const float *efW2 = P[2],  *efb2 = P[3],  *efW3 = P[4],  *efb3 = P[5];
  const float *efg  = P[6],  *efbe = P[7];
  const float *eW2  = P[10], *eb2  = P[11], *eW3  = P[12], *eb3  = P[13];
  const float *eg   = P[14], *ebe  = P[15];
  const float *inW1 = P[16], *inb1 = P[17], *inW2 = P[18], *inb2 = P[19];
  const float *inW3 = P[20], *inb3 = P[21], *ing  = P[22], *inbe = P[23];
  const float *i1W1 = P[24], *i1b1 = P[25], *i1W2 = P[26], *i1b2 = P[27];
  const float *i1W3 = P[28], *i1b3 = P[29];
  const float *mW1  = P[30], *mb1  = P[31], *mW2  = P[32], *mb2  = P[33];
  const float *mW3  = P[34], *mb3  = P[35];
  const float *fW1  = P[36], *fb1  = P[37], *fW2  = P[38], *fb2  = P[39];
  const float *fW3  = P[40], *fb3  = P[41];

  // ---- workspace carve-up ----
  char* ws = (char*)d_ws;
  const size_t U = (size_t)N_EDGES * 128 * sizeof(_Float16);  // 102.4 MB
  _Float16* A   = (_Float16*)(ws + 0 * U);
  _Float16* B   = (_Float16*)(ws + 1 * U);
  _Float16* C   = (_Float16*)(ws + 2 * U);
  _Float16* D   = (_Float16*)(ws + 3 * U);
  _Float16* MSG = (_Float16*)(ws + 4 * U);                    // [E,384] = 3U
  float*    FIJ = (float*)   (ws + 7 * U);                    // [N,3]
  char* nbase   = ws + 7 * U + (size_t)(4 << 20);
  const size_t NU = (size_t)N_NODES * 128 * sizeof(_Float16); // 25.6 MB
  _Float16* NL16 = (_Float16*)(nbase + 0 * NU);
  _Float16* ND   = (_Float16*)(nbase + 1 * NU);
  _Float16* HM   = (_Float16*)(nbase + 2 * NU);
  _Float16* HF   = (_Float16*)(nbase + 3 * NU);

  // transposed f16 weights (one-time prep)
  _Float16* WT = (_Float16*)(nbase + 4 * NU);
  const size_t W128 = 128 * 128;
  _Float16* tEfW2 = WT;                 _Float16* tEfW3 = tEfW2 + W128;
  _Float16* tEW2  = tEfW3 + W128;       _Float16* tEW3  = tEW2  + W128;
  _Float16* tInW1 = tEW3  + W128;       // 128*384
  _Float16* tInW2 = tInW1 + 128 * 384;  _Float16* tInW3 = tInW2 + W128;
  _Float16* tI1W1 = tInW3 + W128;       _Float16* tI1W2 = tI1W1 + W128;
  _Float16* tMW1  = tI1W2 + W128;       _Float16* tMW2  = tMW1  + W128;
  _Float16* tFW1  = tMW2  + W128;       _Float16* tFW2  = tFW1  + W128;

  float* out = (float*)d_out;
  float* resid_out = out;                        // [N,3]
  float* inter_out = out + (size_t)N_NODES * 3;  // [E,128]

  const int EB  = (N_EDGES + 255) / 256;          // 1563
  const int EG  = (N_EDGES + 127) / 128;          // 3125 gemm blocks (edges)
  const int NG  = (N_NODES + 127) / 128;          // 782  gemm blocks (nodes)
  const long long EL = (long long)N_EDGES * 128;
  const int MB  = (int)((EL + 255) / 256);        // msg blocks
  const long long NL = (long long)N_NODES * 128;

  auto prep = [&](const float* Wsrc, _Float16* Wt, int K) {
    int total = 128 * K;
    wprep_kernel<<<(total + 255) / 256, 256, 0, stream>>>(Wsrc, Wt, K);
  };
  // one-time transposed f16 weight prep
  prep(efW2, tEfW2, 128);  prep(efW3, tEfW3, 128);
  prep(eW2,  tEW2,  128);  prep(eW3,  tEW3,  128);
  prep(inW1, tInW1, 384);  prep(inW2, tInW2, 128);  prep(inW3, tInW3, 128);
  prep(i1W1, tI1W1, 128);  prep(i1W2, tI1W2, 128);
  prep(mW1,  tMW1,  128);  prep(mW2,  tMW2,  128);
  prep(fW1,  tFW1,  128);  prep(fW2,  tFW2,  128);

  #define GEMM_RELU(src, Wt, b, dst, M, G) \
    gemm128_kernel<128, true, false, true, false><<<G, 256, 0, stream>>>( \
        src, Wt, b, nullptr, nullptr, dst, nullptr, M)
  #define GEMM_LN(src, Wt, b, g, be, dst, M, G) \
    gemm128_kernel<128, false, true, true, false><<<G, 256, 0, stream>>>( \
        src, Wt, b, g, be, dst, nullptr, M)

  // 1. edge front (features + first layers) -> A=hs1, B=hr1, C=he1
  edge_front_kernel<<<EB, 256, 0, stream>>>(
      edge_dx, edge_attr, va, vb, vc, svt, svtm1, rvt, rvtm1,
      P[0], P[1], P[8], P[9], A, B, C);

  // 2-7. edge_feat / edge MLP tails (WMMA)
  GEMM_RELU(A, tEfW2, efb2, D, N_EDGES, EG);
  GEMM_LN  (D, tEfW3, efb3, efg, efbe, A, N_EDGES, EG);   // A = s_lat
  GEMM_RELU(B, tEfW2, efb2, D, N_EDGES, EG);
  GEMM_LN  (D, tEfW3, efb3, efg, efbe, B, N_EDGES, EG);   // B = r_lat
  GEMM_RELU(C, tEW2, eb2, D, N_EDGES, EG);
  GEMM_LN  (D, tEW3, eb3, eg, ebe, C, N_EDGES, EG);       // C = e_lat

  // 8. msg_input [E,384]
  msg_kernel<<<MB, 256, 0, stream>>>(A, B, C, node_latent, edge_index, MSG);

  // 9-11. inter MLP (K=384 first layer); layer3 LN -> f32 output + f16 copy
  gemm128_kernel<384, true, false, true, false><<<EG, 256, 0, stream>>>(
      MSG, tInW1, inb1, nullptr, nullptr, A, nullptr, N_EDGES);
  GEMM_RELU(A, tInW2, inb2, B, N_EDGES, EG);
  gemm128_kernel<128, false, true, true, true><<<EG, 256, 0, stream>>>(
      B, tInW3, inb3, ing, inbe, C, inter_out, N_EDGES);  // C = interaction (f16)

  // 12-13. i1 hidden layers
  GEMM_RELU(C, tI1W1, i1b1, A, N_EDGES, EG);
  GEMM_RELU(A, tI1W2, i1b2, B, N_EDGES, EG);

  // 14-15. fij scatter
  zero_kernel<<<(N_NODES * 3 + 255) / 256, 256, 0, stream>>>(FIJ, (long long)N_NODES * 3);
  fij_scatter_kernel<<<EB, 256, 0, stream>>>(B, i1W3, i1b3, va, vb, vc,
                                             edge_index, FIJ);

  // 16-20. node path
  cvt16_kernel<<<(int)((NL + 255) / 256), 256, 0, stream>>>(node_latent, NL16, NL);
  GEMM_RELU(NL16, tMW1, mb1, ND, N_NODES, NG);
  GEMM_RELU(ND,   tMW2, mb2, HM, N_NODES, NG);
  GEMM_RELU(NL16, tFW1, fb1, ND, N_NODES, NG);
  GEMM_RELU(ND,   tFW2, fb2, HF, N_NODES, NG);

  // 21. residual
  node_final_kernel<<<(N_NODES + 255) / 256, 256, 0, stream>>>(
      HM, HF, mW3, mb3, fW3, fb3, node_acc, FIJ, resid_out);

  #undef GEMM_RELU
  #undef GEMM_LN
}
